// BeliefTransformerBlock_36919538876735
// MI455X (gfx1250) — matrix operand order repaired and checked by reading, not compiled, not run on hardware
//
#include <hip/hip_runtime.h>
#include <hip/hip_bf16.h>

// ---------------------------------------------------------------------------
// BeliefTransformerBlock for MI455X (gfx1250, wave32, WMMA).
// All GEMMs run as bf16 x bf16 -> f32 via v_wmma_f32_16x16x32_bf16.
// LayerNorm / softmax / GELU / residuals stay in f32.
// GEMM global->LDS staging uses GLOBAL_LOAD_ASYNC_TO_LDS_B128 (ASYNCcnt)
// when the toolchain exposes the builtin; otherwise reg-staged fallback.
// ---------------------------------------------------------------------------

#define Bsz  8
#define Ssz  1024
#define Dsz  1024
#define Hsz  8
#define HDsz 128
#define NEGV  (-1e10f)
#define SCALE 0.08838834764831845f   // 1/sqrt(128)

typedef __bf16 bf16_t;
typedef __bf16 v16bf __attribute__((ext_vector_type(16)));
typedef float  v8f   __attribute__((ext_vector_type(8)));

#define LDSP 40   // padded LDS row stride (elements) for 32-wide K tiles

union Frag  { uint4 u[2]; v16bf v; };
union BF16x { uint4 u[2]; bf16_t h[16]; };

// -------- async global->LDS probe (compile-safe) ---------------------------
#if defined(__has_builtin)
#  if __has_builtin(__builtin_amdgcn_global_load_async_to_lds_b128) && \
      __has_builtin(__builtin_amdgcn_s_wait_asynccnt)
#    define USE_ASYNC_LDS 1
#  endif
#endif
#ifndef USE_ASYNC_LDS
#  define USE_ASYNC_LDS 0
#endif

#if USE_ASYNC_LDS
// Prototype (from hipcc diagnostic): parameters are pointers to
// 'int __attribute__((vector_size(16)))' in AS(1) (global) / AS(3) (LDS).
typedef int v4i_gv __attribute__((vector_size(16)));

__device__ __forceinline__ void async_cp128(const bf16_t* g, bf16_t* l) {
  __builtin_amdgcn_global_load_async_to_lds_b128(
      (__attribute__((address_space(1))) v4i_gv*)g,
      (__attribute__((address_space(3))) v4i_gv*)l,
      /*offset=*/0, /*cpol=*/0);
}
#endif

// ---------------------------------------------------------------------------
// f32 -> bf16 bulk convert (weights)
// ---------------------------------------------------------------------------
__global__ __launch_bounds__(256) void cvt_bf16(const float* __restrict__ in,
                                                bf16_t* __restrict__ out, int n4) {
  int i = blockIdx.x * blockDim.x + threadIdx.x;
  if (i < n4) {
    float4 v = reinterpret_cast<const float4*>(in)[i];
    union { bf16_t h[4]; uint2 u; } pk;
    pk.h[0] = (bf16_t)v.x; pk.h[1] = (bf16_t)v.y;
    pk.h[2] = (bf16_t)v.z; pk.h[3] = (bf16_t)v.w;
    reinterpret_cast<uint2*>(out)[i] = pk.u;
  }
}

// ---------------------------------------------------------------------------
// LayerNorm: one row (D=1024) per block, 256 threads, 4 elems/thread.
// Writes bf16 always; optional f32 copy (may alias input: reads precede writes).
// ---------------------------------------------------------------------------
__global__ __launch_bounds__(256) void ln_kernel(const float* __restrict__ x,
                                                 const float* __restrict__ w,
                                                 bf16_t* __restrict__ ob,
                                                 float* __restrict__ of) {
  __shared__ float red[2][8];
  const int tid = threadIdx.x, wid = tid >> 5, lane = tid & 31;
  const size_t row = blockIdx.x;
  const float4 v = reinterpret_cast<const float4*>(x + row * Dsz)[tid];

  float s = v.x + v.y + v.z + v.w;
  float q = v.x * v.x + v.y * v.y + v.z * v.z + v.w * v.w;
#pragma unroll
  for (int o = 16; o > 0; o >>= 1) { s += __shfl_xor(s, o, 32); q += __shfl_xor(q, o, 32); }
  if (!lane) { red[0][wid] = s; red[1][wid] = q; }
  __syncthreads();
  if (tid < 32) {
    s = (lane < 8) ? red[0][lane] : 0.f;
    q = (lane < 8) ? red[1][lane] : 0.f;
#pragma unroll
    for (int o = 4; o > 0; o >>= 1) { s += __shfl_xor(s, o, 32); q += __shfl_xor(q, o, 32); }
    if (!lane) { red[0][0] = s; red[1][0] = q; }
  }
  __syncthreads();
  const float mean = red[0][0] * (1.f / Dsz);
  const float var  = red[1][0] * (1.f / Dsz) - mean * mean;
  const float rs   = rsqrtf(var + 1e-5f);
  const float4 wv  = reinterpret_cast<const float4*>(w)[tid];

  float4 o;
  o.x = (v.x - mean) * rs * wv.x;
  o.y = (v.y - mean) * rs * wv.y;
  o.z = (v.z - mean) * rs * wv.z;
  o.w = (v.w - mean) * rs * wv.w;

  union { bf16_t h[4]; uint2 u; } pk;
  pk.h[0] = (bf16_t)o.x; pk.h[1] = (bf16_t)o.y;
  pk.h[2] = (bf16_t)o.z; pk.h[3] = (bf16_t)o.w;
  reinterpret_cast<uint2*>(ob + row * Dsz)[tid] = pk.u;
  if (of) reinterpret_cast<float4*>(of + row * Dsz)[tid] = o;
}

// ---------------------------------------------------------------------------
// In-place softmax over rows of length S=1024 (one row per block).
// ---------------------------------------------------------------------------
__global__ __launch_bounds__(256) void softmax_kernel(float* __restrict__ att) {
  __shared__ float red[8];
  const int tid = threadIdx.x, wid = tid >> 5, lane = tid & 31;
  float* p = att + (size_t)blockIdx.x * Ssz;
  float4 v = reinterpret_cast<float4*>(p)[tid];

  float m = fmaxf(fmaxf(v.x, v.y), fmaxf(v.z, v.w));
#pragma unroll
  for (int o = 16; o > 0; o >>= 1) m = fmaxf(m, __shfl_xor(m, o, 32));
  if (!lane) red[wid] = m;
  __syncthreads();
  const float bm = fmaxf(fmaxf(fmaxf(red[0], red[1]), fmaxf(red[2], red[3])),
                         fmaxf(fmaxf(red[4], red[5]), fmaxf(red[6], red[7])));
  __syncthreads();

  const float e0 = __expf(v.x - bm), e1 = __expf(v.y - bm);
  const float e2 = __expf(v.z - bm), e3 = __expf(v.w - bm);
  float s = e0 + e1 + e2 + e3;
#pragma unroll
  for (int o = 16; o > 0; o >>= 1) s += __shfl_xor(s, o, 32);
  if (!lane) red[wid] = s;
  __syncthreads();
  const float bs  = red[0] + red[1] + red[2] + red[3] + red[4] + red[5] + red[6] + red[7];
  const float inv = 1.f / bs;
  v.x = e0 * inv; v.y = e1 * inv; v.z = e2 * inv; v.w = e3 * inv;
  reinterpret_cast<float4*>(p)[tid] = v;
}

// ---------------------------------------------------------------------------
// Generic C[M,N] = A[M,K] * W[N,K]^T  (bf16 inputs, f32 accumulate via WMMA).
// Block tile 128x128, 8 waves (2x4), wave tile 64x32 = 4x2 WMMA tiles.
// Double-buffered LDS, one barrier per 32-wide K step. Async DMA staging
// when available: wait own ASYNCcnt -> barrier -> prefetch next -> compute.
// ---------------------------------------------------------------------------
enum { M_QKV = 0, M_FC = 1, M_PROJ = 2, M_SCORES = 3 };

template <int MODE>
__global__ __launch_bounds__(256) void gemm_bt(
    const bf16_t* __restrict__ A, int lda,
    const bf16_t* __restrict__ Bw, int ldb,
    void* __restrict__ Cvoid, int ldc,
    int M, int N, int K,
    const float* __restrict__ resid,
    const int* __restrict__ sizes) {
  (void)M; (void)N;
  __shared__ __align__(16) bf16_t sA[2][128 * LDSP];
  __shared__ __align__(16) bf16_t sB[2][128 * LDSP];

  const int tid = threadIdx.x;
  const int wid = tid >> 5, lane = tid & 31, hs = lane >> 4, l15 = lane & 15;
  const int bm = blockIdx.y * 128, bn = blockIdx.x * 128;

  int cursize = 0;
  size_t coutOff = 0;
  if constexpr (MODE == M_SCORES) {
    const int z = blockIdx.z, b = z >> 3, h = z & 7;
    const size_t qoff = (size_t)b * Ssz * (3 * Dsz) + (size_t)h * HDsz;
    A += qoff;           // q head slice
    Bw += qoff;          // k head slice (base already offset by +D)
    coutOff = (size_t)z * Ssz * Ssz;
    cursize = sizes[b];
  }
  float*  outF = (float*)Cvoid + coutOff;
  bf16_t* outB = (bf16_t*)Cvoid;

  // Global->LDS staging: thread t covers 32B of one tile row (2 threads/row).
  const int ldRow = tid >> 1;
  const int ldCol = (tid & 1) * 16;

  v8f acc[4][2] = {};
  const int waveM = (wid >> 2) * 64;
  const int waveN = (wid & 3) * 32;

#if USE_ASYNC_LDS
  auto aload = [&](int kk, int bb) {
    const bf16_t* ag = A + (size_t)(bm + ldRow) * lda + kk + ldCol;
    bf16_t* al = &sA[bb][ldRow * LDSP + ldCol];
    async_cp128(ag, al);
    async_cp128(ag + 8, al + 8);
    const bf16_t* bg = Bw + (size_t)(bn + ldRow) * ldb + kk + ldCol;
    bf16_t* bl = &sB[bb][ldRow * LDSP + ldCol];
    async_cp128(bg, bl);
    async_cp128(bg + 8, bl + 8);
  };
  aload(0, 0);
#else
  uint4 ra0, ra1, rb0, rb1;
  auto gload = [&](int kk) {
    const uint4* a4 = reinterpret_cast<const uint4*>(A + (size_t)(bm + ldRow) * lda + kk + ldCol);
    ra0 = a4[0]; ra1 = a4[1];
    const uint4* b4 = reinterpret_cast<const uint4*>(Bw + (size_t)(bn + ldRow) * ldb + kk + ldCol);
    rb0 = b4[0]; rb1 = b4[1];
  };
  gload(0);
#endif

  int buf = 0;
  for (int kk = 0; kk < K; kk += 32) {
#if USE_ASYNC_LDS
    __builtin_amdgcn_s_wait_asynccnt(0);  // this wave's DMA into sX[buf] done
    __syncthreads();                      // all waves' DMA done + prior compute done
    if (kk + 32 < K) aload(kk + 32, buf ^ 1);  // DMA overlaps compute below
#else
    {
      uint4* pa = reinterpret_cast<uint4*>(&sA[buf][ldRow * LDSP + ldCol]);
      pa[0] = ra0; pa[1] = ra1;
      uint4* pb = reinterpret_cast<uint4*>(&sB[buf][ldRow * LDSP + ldCol]);
      pb[0] = rb0; pb[1] = rb1;
    }
    __syncthreads();
    if (kk + 32 < K) gload(kk + 32);
#endif
    // Warm GL2 two K-steps ahead (global_prefetch_b8).
    if (kk + 64 < K) {
      __builtin_prefetch(A + (size_t)(bm + ldRow) * lda + kk + 64 + ldCol, 0, 2);
      __builtin_prefetch(Bw + (size_t)(bn + ldRow) * ldb + kk + 64 + ldCol, 0, 2);
    }

    // A fragments: lane<16 -> K {0..7,16..23}; lane>=16 -> K {8..15,24..31}
    Frag af[4], bfr[2];
#pragma unroll
    for (int mi = 0; mi < 4; ++mi) {
      const bf16_t* p = &sA[buf][(waveM + mi * 16 + l15) * LDSP + hs * 8];
      af[mi].u[0] = *reinterpret_cast<const uint4*>(p);
      af[mi].u[1] = *reinterpret_cast<const uint4*>(p + 16);
    }
    // B fragments: lane<16 -> K {0..7,16..23}; lane>=16 -> K {8..15,24..31}
#pragma unroll
    for (int ni = 0; ni < 2; ++ni) {
      const bf16_t* p = &sB[buf][(waveN + ni * 16 + l15) * LDSP + hs * 8];
      bfr[ni].u[0] = *reinterpret_cast<const uint4*>(p);
      bfr[ni].u[1] = *reinterpret_cast<const uint4*>(p + 16);
    }
#pragma unroll
    for (int mi = 0; mi < 4; ++mi)
#pragma unroll
      for (int ni = 0; ni < 2; ++ni)
        acc[mi][ni] = __builtin_amdgcn_wmma_f32_16x16x32_bf16(
            false, af[mi].v, false, bfr[ni].v, (short)0, acc[mi][ni], false, false);
    buf ^= 1;
  }

  // Epilogue. C layout: VGPR r holds M = r + hs*8, N = lane&15.
#pragma unroll
  for (int mi = 0; mi < 4; ++mi) {
#pragma unroll
    for (int ni = 0; ni < 2; ++ni) {
      const int col  = bn + waveN + ni * 16 + l15;
      const int row0 = bm + waveM + mi * 16 + hs * 8;
#pragma unroll
      for (int r = 0; r < 8; ++r) {
        const int row = row0 + r;
        const float val = acc[mi][ni][r];
        if constexpr (MODE == M_QKV) {
          outB[(size_t)row * ldc + col] = (bf16_t)val;
        } else if constexpr (MODE == M_FC) {
          const float g = 0.5f * val * (1.f + erff(val * 0.70710678118654752f));
          outB[(size_t)row * ldc + col] = (bf16_t)g;
        } else if constexpr (MODE == M_PROJ) {
          const size_t idx = (size_t)row * ldc + col;
          outF[idx] = val + resid[idx];
        } else {  // M_SCORES: scale + key-padding mask
          outF[(size_t)row * ldc + col] = (col >= cursize) ? NEGV : val * SCALE;
        }
      }
    }
  }
}

// ---------------------------------------------------------------------------
// y = att @ v (per b,h), plus residual old_x, written f32.
// A = att (f32, converted to bf16 on the fly); B = v rows transposed into
// the WMMA B layout in LDS. M=S=1024, N=HD=128, K=S=1024.
// ---------------------------------------------------------------------------
__global__ __launch_bounds__(256) void attn_av_kernel(
    const float* __restrict__ att, const bf16_t* __restrict__ qkv,
    const float* __restrict__ xin, float* __restrict__ resid) {
  __shared__ __align__(16) bf16_t sA[2][128 * LDSP];
  __shared__ __align__(16) bf16_t sB[2][128 * LDSP];

  const int tid = threadIdx.x;
  const int wid = tid >> 5, lane = tid & 31, hs = lane >> 4, l15 = lane & 15;
  const int z = blockIdx.z, b = z >> 3, h = z & 7;
  const int bm = blockIdx.y * 128;

  const float*  aBase = att + (size_t)z * Ssz * Ssz;
  const bf16_t* vBase = qkv + (size_t)b * Ssz * (3 * Dsz) + 2 * Dsz + (size_t)h * HDsz;

  const int aRow = tid >> 1, aCol = (tid & 1) * 16;      // att: 16 f32 / thread
  const int kRow = tid >> 3, dSeg = (tid & 7) * 16;      // v: 16 bf16 / thread

  float4 raf[4];
  BF16x  rbv;
  auto gload = [&](int kk) {
    const float4* ap = reinterpret_cast<const float4*>(aBase + (size_t)(bm + aRow) * Ssz + kk + aCol);
    raf[0] = ap[0]; raf[1] = ap[1]; raf[2] = ap[2]; raf[3] = ap[3];
    const uint4* bp = reinterpret_cast<const uint4*>(vBase + (size_t)(kk + kRow) * (3 * Dsz) + dSeg);
    rbv.u[0] = bp[0]; rbv.u[1] = bp[1];
  };
  auto sstore = [&](int bb) {
    BF16x pk;
#pragma unroll
    for (int i = 0; i < 4; ++i) {
      pk.h[i * 4 + 0] = (bf16_t)raf[i].x; pk.h[i * 4 + 1] = (bf16_t)raf[i].y;
      pk.h[i * 4 + 2] = (bf16_t)raf[i].z; pk.h[i * 4 + 3] = (bf16_t)raf[i].w;
    }
    uint4* pa = reinterpret_cast<uint4*>(&sA[bb][aRow * LDSP + aCol]);
    pa[0] = pk.u[0]; pa[1] = pk.u[1];
    // transpose v tile: sB[d][k]
#pragma unroll
    for (int j = 0; j < 16; ++j) sB[bb][(dSeg + j) * LDSP + kRow] = rbv.h[j];
  };

  v8f acc[4][2] = {};
  const int waveM = (wid >> 2) * 64;
  const int waveN = (wid & 3) * 32;

  gload(0);
  int buf = 0;
  for (int kk = 0; kk < Ssz; kk += 32) {
    sstore(buf);
    __syncthreads();
    if (kk + 32 < Ssz) gload(kk + 32);

    Frag af[4], bfr[2];
#pragma unroll
    for (int mi = 0; mi < 4; ++mi) {
      const bf16_t* p = &sA[buf][(waveM + mi * 16 + l15) * LDSP + hs * 8];
      af[mi].u[0] = *reinterpret_cast<const uint4*>(p);
      af[mi].u[1] = *reinterpret_cast<const uint4*>(p + 16);
    }
#pragma unroll
    for (int ni = 0; ni < 2; ++ni) {
      const bf16_t* p = &sB[buf][(waveN + ni * 16 + l15) * LDSP + hs * 8];
      bfr[ni].u[0] = *reinterpret_cast<const uint4*>(p);
      bfr[ni].u[1] = *reinterpret_cast<const uint4*>(p + 16);
    }
#pragma unroll
    for (int mi = 0; mi < 4; ++mi)
#pragma unroll
      for (int ni = 0; ni < 2; ++ni)
        acc[mi][ni] = __builtin_amdgcn_wmma_f32_16x16x32_bf16(
            false, af[mi].v, false, bfr[ni].v, (short)0, acc[mi][ni], false, false);
    buf ^= 1;
  }

#pragma unroll
  for (int mi = 0; mi < 4; ++mi) {
#pragma unroll
    for (int ni = 0; ni < 2; ++ni) {
      const int col  = waveN + ni * 16 + l15;          // d within head
      const int row0 = bm + waveM + mi * 16 + hs * 8;  // s
#pragma unroll
      for (int r = 0; r < 8; ++r) {
        const int row = row0 + r;
        const size_t idx = ((size_t)b * Ssz + row) * Dsz + (size_t)h * HDsz + col;
        resid[idx] = acc[mi][ni][r] + xin[idx];
      }
    }
  }
}

// ---------------------------------------------------------------------------
// Host-side orchestration (graph-capture safe: launches only).
// ---------------------------------------------------------------------------
extern "C" void kernel_launch(void* const* d_in, const int* in_sizes, int n_in,
                              void* d_out, int out_size, void* d_ws, size_t ws_size,
                              hipStream_t stream) {
  (void)in_sizes; (void)n_in; (void)out_size; (void)ws_size;

  const float* x     = (const float*)d_in[0];
  const int*   sizes = (const int*)  d_in[1];
  const float* ln1w  = (const float*)d_in[2];
  const float* qkvw  = (const float*)d_in[3];
  const float* ln2w  = (const float*)d_in[4];
  const float* fcw   = (const float*)d_in[5];
  const float* projw = (const float*)d_in[6];

  float* out = (float*)d_out;                       // [B,S,D] final
  float* att = out + (size_t)Bsz * Ssz * Dsz;       // [B,H,S,S] attention

  char* ws = (char*)d_ws;
  bf16_t* qkvw_b  = (bf16_t*)(ws + 0);              //  6.0 MB
  bf16_t* fcw_b   = (bf16_t*)(ws + 6291456);        //  2.0 MB
  bf16_t* projw_b = (bf16_t*)(ws + 8388608);        //  2.0 MB
  bf16_t* xn      = (bf16_t*)(ws + 10485760);       // 16.0 MB (reused as h)
  bf16_t* qkvact  = (bf16_t*)(ws + 27262976);       // 48.0 MB
  float*  resid   = (float*) (ws + 77594624);       // 32.0 MB (reused as x2 f32)
  bf16_t* x2b     = (bf16_t*)(ws + 111149056);      // 16.0 MB

  // Weights -> bf16
  cvt_bf16<<<3072, 256, 0, stream>>>(qkvw,  qkvw_b,  3 * 1024 * 1024 / 4);
  cvt_bf16<<<1024, 256, 0, stream>>>(fcw,   fcw_b,   1024 * 1024 / 4);
  cvt_bf16<<<1024, 256, 0, stream>>>(projw, projw_b, 1024 * 1024 / 4);

  // xn = LN1(x) -> bf16
  ln_kernel<<<Bsz * Ssz, 256, 0, stream>>>(x, ln1w, xn, nullptr);

  // qkv = xn @ qkv_w^T   [8192 x 3072]
  gemm_bt<M_QKV><<<dim3(24, 64, 1), 256, 0, stream>>>(
      xn, Dsz, qkvw_b, Dsz, qkvact, 3 * Dsz, Bsz * Ssz, 3 * Dsz, Dsz, nullptr, nullptr);

  // scores = scale * q k^T with key-padding mask, written into att region
  gemm_bt<M_SCORES><<<dim3(8, 8, Bsz * Hsz), 256, 0, stream>>>(
      qkvact, 3 * Dsz, qkvact + Dsz, 3 * Dsz, att, Ssz, Ssz, Ssz, HDsz, nullptr, sizes);

  // softmax in place over att rows
  softmax_kernel<<<Bsz * Hsz * Ssz, 256, 0, stream>>>(att);

  // resid = att @ v + x
  attn_av_kernel<<<dim3(1, 8, Bsz * Hsz), 256, 0, stream>>>(att, qkvact, x, resid);

  // x2 = LN2(resid): bf16 for GEMM, f32 (in place over resid) for final add
  ln_kernel<<<Bsz * Ssz, 256, 0, stream>>>(resid, ln2w, x2b, resid);

  // h = gelu(x2 @ fc_w^T) -> bf16 (reuse xn buffer)
  gemm_bt<M_FC><<<dim3(8, 64, 1), 256, 0, stream>>>(
      x2b, Dsz, fcw_b, Dsz, xn, Dsz, Bsz * Ssz, Dsz, Dsz, nullptr, nullptr);

  // out = h @ proj_w^T + x2
  gemm_bt<M_PROJ><<<dim3(8, 64, 1), 256, 0, stream>>>(
      xn, Dsz, projw_b, Dsz, out, Dsz, Bsz * Ssz, Dsz, Dsz, resid, nullptr);
}
